// Intent_Slot_Attention_36558761624275
// MI455X (gfx1250) — matrix operand order repaired and checked by reading, not compiled, not run on hardware
//
#include <hip/hip_runtime.h>
#include <hip/hip_bf16.h>
#include <stdint.h>

typedef __bf16 v16bf __attribute__((ext_vector_type(16)));
typedef __bf16 v8bf  __attribute__((ext_vector_type(8)));
typedef float  v8f   __attribute__((ext_vector_type(8)));
typedef int    v4i   __attribute__((ext_vector_type(4)));

typedef v4i __attribute__((address_space(1)))* gptr4i;   // global <4 x i32>*
typedef v4i __attribute__((address_space(3)))* lptr4i;   // LDS    <4 x i32>*

#define HDIM  1024
#define BATCH 8
#define SEQ   4096
#define LSEQ  64

#if __has_builtin(__builtin_amdgcn_global_load_async_to_lds_b128)
#define ASYNC_STAGE 1
#else
#define ASYNC_STAGE 0
#endif

// ---------------- WMMA helpers (CDNA5 16x16x32 bf16) ----------------

__device__ inline v8f wmma_bf16(v16bf a, v16bf b, v8f c) {
  // (neg_a, A, neg_b, B, c_mod, C, reuse_a, reuse_b)
  return __builtin_amdgcn_wmma_f32_16x16x32_bf16(false, a, false, b, (short)0, c,
                                                 false, false);
}

__device__ inline v8f zero8() {
  v8f z;
#pragma unroll
  for (int i = 0; i < 8; ++i) z[i] = 0.0f;
  return z;
}

// A fragment (16x32, MxK). Lane: m = lane%16, half h = lane/16.
// element e<8 -> K = k0+8h+e ; e>=8 -> K = k0+16+8h+(e-8).
__device__ inline v16bf load_a_f32(const float* rb, int k0, int hh) {
  float4 f[4];
  const float* s1 = rb + k0 + 8 * hh;
  f[0] = *(const float4*)(s1);
  f[1] = *(const float4*)(s1 + 4);
  f[2] = *(const float4*)(s1 + 16);
  f[3] = *(const float4*)(s1 + 20);
  const float* tf = (const float*)f;
  v16bf a;
#pragma unroll
  for (int e = 0; e < 16; ++e) a[e] = (__bf16)tf[e];
  return a;
}

__device__ inline v16bf load_a_bf16(const __bf16* rb, int k0, int hh) {
  const v8bf* s1 = (const v8bf*)(rb + k0 + 8 * hh);
  const v8bf* s2 = (const v8bf*)(rb + k0 + 8 * hh + 16);
  v8bf lo = *s1, hi = *s2;
  v16bf a;
#pragma unroll
  for (int e = 0; e < 8; ++e) { a[e] = lo[e]; a[8 + e] = hi[e]; }
  return a;
}

// B fragment (32x16, KxN). Lane: n = lane%16, half h: e<8 -> K = k0+8h+e ; e>=8 -> K = k0+16+8h+(e-8).
// Source is NxK row-major (contiguous K); rb points at this lane's N row.
__device__ inline v16bf load_b_f32(const float* rb, int k0, int hh) {
  float4 f[4];
  const float* s = rb + k0 + 8 * hh;
  f[0] = *(const float4*)(s);
  f[1] = *(const float4*)(s + 4);
  f[2] = *(const float4*)(s + 16);
  f[3] = *(const float4*)(s + 20);
  const float* tf = (const float*)f;
  v16bf b;
#pragma unroll
  for (int e = 0; e < 16; ++e) b[e] = (__bf16)tf[e];
  return b;
}

__device__ inline v16bf load_b_bf16(const __bf16* rb, int k0, int hh) {
  const v8bf* s = (const v8bf*)(rb + k0 + 8 * hh);
  v8bf lo = s[0], hi = s[2];
  v16bf b;
#pragma unroll
  for (int e = 0; e < 8; ++e) { b[e] = lo[e]; b[8 + e] = hi[e]; }
  return b;
}

// 16-byte global -> LDS copy; async (ASYNCcnt) when the builtin exists.
// Builtin prototype is (v4i AS1*, v4i AS3*, imm offset, imm cpol); go through
// integers for the address-space casts (low 32 bits of a generic LDS pointer
// are the AS3 offset per the flat-aperture rules).
__device__ inline void copy16_g2l(const __bf16* g, __bf16* l) {
#if ASYNC_STAGE
  gptr4i gp = (gptr4i)(uintptr_t)g;
  lptr4i lp = (lptr4i)(uint32_t)(uintptr_t)l;
  __builtin_amdgcn_global_load_async_to_lds_b128(gp, lp, 0, 0);
#else
  *(v8bf*)l = *(const v8bf*)g;
#endif
}

__device__ inline void wait_async_stage() {
#if ASYNC_STAGE
#if __has_builtin(__builtin_amdgcn_s_wait_asynccnt)
  __builtin_amdgcn_s_wait_asynccnt(0);
#else
  asm volatile("s_wait_asynccnt 0" ::: "memory");
#endif
#endif
}

// ---------------- Kernel A1: k = x@Wk^T + bk (f32), vT = (x@Wv^T + bv)^T (bf16)

__global__ __launch_bounds__(256) void kv_kernel(
    const float* __restrict__ x, const float* __restrict__ Wk,
    const float* __restrict__ bk, const float* __restrict__ Wv,
    const float* __restrict__ bv, float* __restrict__ kf,
    __bf16* __restrict__ vT) {
  const int b = blockIdx.y;
  const int nchunk = blockIdx.x;                 // 128-wide N chunk
  const int tid = threadIdx.x;
  const int w = tid >> 5, lane = tid & 31, ll = lane & 15, hh = lane >> 4;
  const int mt = (w & 3) * 16;                   // row tile within 64 rows
  const int nbase = nchunk * 128 + (w >> 2) * 64;

  const float* xrow = x + ((size_t)b * LSEQ + mt + ll) * HDIM;

  v8f ak[4], av[4];
#pragma unroll
  for (int t = 0; t < 4; ++t) { ak[t] = zero8(); av[t] = zero8(); }

  for (int k0 = 0; k0 < HDIM; k0 += 32) {
    v16bf a = load_a_f32(xrow, k0, hh);
#pragma unroll
    for (int t = 0; t < 4; ++t) {
      const int o = nbase + t * 16 + ll;
      v16bf bkv = load_b_f32(Wk + (size_t)o * HDIM, k0, hh);
      v16bf bvv = load_b_f32(Wv + (size_t)o * HDIM, k0, hh);
      ak[t] = wmma_bf16(a, bkv, ak[t]);
      av[t] = wmma_bf16(a, bvv, av[t]);
    }
  }

#pragma unroll
  for (int t = 0; t < 4; ++t) {
    const int o = nbase + t * 16 + ll;
    const float kb = bk[o], vb = bv[o];
    float* kout = kf + ((size_t)b * LSEQ + mt + 8 * hh) * HDIM + o;
    v8bf vpack;
#pragma unroll
    for (int r = 0; r < 8; ++r) {
      kout[(size_t)r * HDIM] = ak[t][r] + kb;     // rows mt+8h+r
      vpack[r] = (__bf16)(av[t][r] + vb);         // l = mt+8h+r contiguous
    }
    *(v8bf*)(vT + ((size_t)b * HDIM + o) * LSEQ + mt + 8 * hh) = vpack;
  }
}

// ---------------- Kernel A2: Ktil = kf @ Wq (bf16 out), contract over Wq 1st dim

__global__ __launch_bounds__(256) void ktil_kernel(
    const float* __restrict__ kf, const float* __restrict__ Wq,
    __bf16* __restrict__ Ktil) {
  const int b = blockIdx.y;
  const int nchunk = blockIdx.x;
  const int tid = threadIdx.x;
  const int w = tid >> 5, lane = tid & 31, ll = lane & 15, hh = lane >> 4;
  const int mt = (w & 3) * 16;
  const int nbase = nchunk * 128 + (w >> 2) * 64;

  const float* krow = kf + ((size_t)b * LSEQ + mt + ll) * HDIM;

  v8f acc[4];
#pragma unroll
  for (int t = 0; t < 4; ++t) acc[t] = zero8();

  for (int k0 = 0; k0 < HDIM; k0 += 32) {
    v16bf a = load_a_f32(krow, k0, hh);
#pragma unroll
    for (int t = 0; t < 4; ++t) {
      const int n = nbase + t * 16 + ll;
      v16bf bb;                                  // B[k=o][n=i] = Wq[o][i]: strided
#pragma unroll
      for (int e = 0; e < 16; ++e)
        bb[e] = (__bf16)Wq[(size_t)(k0 + 8 * hh + e + (e & 8)) * HDIM + n];
      acc[t] = wmma_bf16(a, bb, acc[t]);
    }
  }

#pragma unroll
  for (int t = 0; t < 4; ++t) {
    const int n = nbase + t * 16 + ll;
    __bf16* kout = Ktil + ((size_t)b * LSEQ + mt + 8 * hh) * HDIM + n;
#pragma unroll
    for (int r = 0; r < 8; ++r) kout[(size_t)r * HDIM] = (__bf16)acc[t][r];
  }
}

// ---------------- Kernel A3: c[b][l] = kf[b][l][:] . bq

__global__ __launch_bounds__(64) void cvec_kernel(const float* __restrict__ kf,
                                                  const float* __restrict__ bq,
                                                  float* __restrict__ cvec) {
  const int b = blockIdx.x;
  const int l = threadIdx.x;
  const float* row = kf + ((size_t)b * LSEQ + l) * HDIM;
  float s = 0.0f;
  for (int i = 0; i < HDIM; ++i) s += row[i] * bq[i];
  cvec[b * LSEQ + l] = s;
}

// ---------------- Kernel B: fused scores -> softmax -> AV -> gated residual

#define KPAD 1032   // 64 rows, 2064B stride: 16B aligned, banks spread (step 4)
#define PPAD 72     // 128 rows, 144B stride: 16B aligned, banks spread (step 36)

__global__ __launch_bounds__(256) void fused_attn(
    const float* __restrict__ p, const __bf16* __restrict__ Ktil,
    const __bf16* __restrict__ vT, const float* __restrict__ cvec,
    const float* __restrict__ Wg, const float* __restrict__ bg,
    float* __restrict__ out) {
  __shared__ __bf16 ktil_lds[LSEQ][KPAD];  // 132096 B, staged once per block
  __shared__ __bf16 probs_lds[128][PPAD];  // 18432 B
  __shared__ float  wsum_lds[HDIM];        // WgO + WgP
  __shared__ float  c_lds[LSEQ];
  __shared__ float  pdot_lds[128];         // per-row p.(WgO+WgP)

  const int b = blockIdx.y;
  const int s0 = blockIdx.x * 128;
  const int tid = threadIdx.x;
  const int w = tid >> 5, lane = tid & 31, ll = lane & 15, hh = lane >> 4;

  // ---- stage Ktil (64x1024 bf16) into LDS with async global->LDS DMA ----
  const __bf16* ktb = Ktil + (size_t)b * LSEQ * HDIM;
  {
    // 8192 16-byte chunks, 32 per thread
#pragma unroll 4
    for (int it = 0; it < 32; ++it) {
      const int ci = tid + it * 256;
      const int row = ci >> 7;             // 128 chunks per row
      const int c8 = (ci & 127) * 8;
      copy16_g2l(ktb + (size_t)row * HDIM + c8, &ktil_lds[row][c8]);
    }
  }
  for (int i = tid; i < HDIM; i += 256) wsum_lds[i] = Wg[i] + Wg[HDIM + i];
  if (tid < LSEQ) c_lds[tid] = cvec[b * LSEQ + tid];
  wait_async_stage();
  __syncthreads();

  // ---- GEMM1: scores(128x64) = p_tile @ Ktil^T ; fold in gate pdot ----
  const float* prow = p + ((size_t)b * SEQ + s0 + w * 16 + ll) * HDIM;

  v8f acc[4];
#pragma unroll
  for (int t = 0; t < 4; ++t) acc[t] = zero8();
  float pdot = 0.0f;

  for (int k0 = 0; k0 < HDIM; k0 += 32) {
    v16bf a = load_a_f32(prow, k0, hh);       // lane owns row w*16+ll
    const float* w1 = &wsum_lds[k0 + 8 * hh];
#pragma unroll
    for (int e = 0; e < 8; ++e)
      pdot += (float)a[e] * w1[e] + (float)a[8 + e] * w1[16 + e];
#pragma unroll
    for (int t = 0; t < 4; ++t) {
      v16bf bb = load_b_bf16(&ktil_lds[t * 16 + ll][0], k0, hh);
      acc[t] = wmma_bf16(a, bb, acc[t]);
    }
  }
  pdot += __shfl_xor(pdot, 16, 32);           // combine the two K halves
  if (hh == 0) pdot_lds[w * 16 + ll] = pdot;

  // ---- +c, softmax over l (row spans 16 lanes of a half x 4 accs) ----
  float mx[8];
#pragma unroll
  for (int r = 0; r < 8; ++r) mx[r] = -1e30f;
#pragma unroll
  for (int t = 0; t < 4; ++t) {
    const float cv = c_lds[t * 16 + ll];
#pragma unroll
    for (int r = 0; r < 8; ++r) {
      acc[t][r] += cv;
      mx[r] = fmaxf(mx[r], acc[t][r]);
    }
  }
#pragma unroll
  for (int m = 1; m < 16; m <<= 1)
#pragma unroll
    for (int r = 0; r < 8; ++r) mx[r] = fmaxf(mx[r], __shfl_xor(mx[r], m, 32));

  float sm[8];
#pragma unroll
  for (int r = 0; r < 8; ++r) sm[r] = 0.0f;
#pragma unroll
  for (int t = 0; t < 4; ++t)
#pragma unroll
    for (int r = 0; r < 8; ++r) {
      acc[t][r] = __expf(acc[t][r] - mx[r]);
      sm[r] += acc[t][r];
    }
#pragma unroll
  for (int m = 1; m < 16; m <<= 1)
#pragma unroll
    for (int r = 0; r < 8; ++r) sm[r] += __shfl_xor(sm[r], m, 32);

  float inv[8];
#pragma unroll
  for (int r = 0; r < 8; ++r) inv[r] = 1.0f / (sm[r] * 32.0f);  // /sqrt(H) post-softmax

#pragma unroll
  for (int t = 0; t < 4; ++t)
#pragma unroll
    for (int r = 0; r < 8; ++r)
      probs_lds[w * 16 + r + 8 * hh][t * 16 + ll] = (__bf16)(acc[t][r] * inv[r]);
  __syncthreads();

  // ---- GEMM2: scoresv = probs(16x64 per wave) @ v(64x1024) ----
  const __bf16* prl = &probs_lds[w * 16 + ll][0];
  v16bf a0 = load_a_bf16(prl, 0, hh);
  v16bf a1 = load_a_bf16(prl, 32, hh);
  const __bf16* vtb = vT + (size_t)b * HDIM * LSEQ;

  // pass 1: gate accumulation  g_row += scoresv . WgO
  float gacc[8];
#pragma unroll
  for (int r = 0; r < 8; ++r) gacc[r] = 0.0f;

  for (int nt = 0; nt < 64; ++nt) {
    const int o0 = nt * 16;
    v8f s = zero8();
    v16bf b0 = load_b_bf16(vtb + (size_t)(o0 + ll) * LSEQ, 0, hh);
    v16bf b1 = load_b_bf16(vtb + (size_t)(o0 + ll) * LSEQ, 32, hh);
    s = wmma_bf16(a0, b0, s);
    s = wmma_bf16(a1, b1, s);
    const float wg = Wg[o0 + ll];
#pragma unroll
    for (int r = 0; r < 8; ++r) gacc[r] += s[r] * wg;
    // warm L0/L2 for the pass-2 residual reads
    __builtin_prefetch(p + ((size_t)b * SEQ + s0 + w * 16 + 8 * hh) * HDIM + o0 + ll,
                       0, 1);
  }
#pragma unroll
  for (int m = 1; m < 16; m <<= 1)
#pragma unroll
    for (int r = 0; r < 8; ++r) gacc[r] += __shfl_xor(gacc[r], m, 32);

  const float bgv = bg[0];
  float g[8];
#pragma unroll
  for (int r = 0; r < 8; ++r) {
    const float tt = gacc[r] + pdot_lds[w * 16 + r + 8 * hh] + bgv;
    g[r] = 1.0f / (1.0f + __expf(-tt));
  }

  // pass 2: recompute scoresv, blend with residual, store
  for (int nt = 0; nt < 64; ++nt) {
    const int o0 = nt * 16;
    v8f s = zero8();
    v16bf b0 = load_b_bf16(vtb + (size_t)(o0 + ll) * LSEQ, 0, hh);
    v16bf b1 = load_b_bf16(vtb + (size_t)(o0 + ll) * LSEQ, 32, hh);
    s = wmma_bf16(a0, b0, s);
    s = wmma_bf16(a1, b1, s);
    const size_t base = ((size_t)b * SEQ + s0 + w * 16 + 8 * hh) * HDIM + o0 + ll;
    const float* pc = p + base;
    float* oc = out + base;
#pragma unroll
    for (int r = 0; r < 8; ++r) {
      const float pv = pc[(size_t)r * HDIM];
      // final = g*out + (1-g)*p = p + g*(out - p) = p + g*scoresv
      oc[(size_t)r * HDIM] = pv + g[r] * s[r];
    }
  }
}

// ---------------- launch ----------------

extern "C" void kernel_launch(void* const* d_in, const int* in_sizes, int n_in,
                              void* d_out, int out_size, void* d_ws, size_t ws_size,
                              hipStream_t stream) {
  const float* p  = (const float*)d_in[0];
  const float* x  = (const float*)d_in[1];
  const float* Wq = (const float*)d_in[2];
  const float* bq = (const float*)d_in[3];
  const float* Wk = (const float*)d_in[4];
  const float* bk = (const float*)d_in[5];
  const float* Wv = (const float*)d_in[6];
  const float* bv = (const float*)d_in[7];
  const float* Wg = (const float*)d_in[8];
  const float* bg = (const float*)d_in[9];
  float* out = (float*)d_out;

  char* ws = (char*)d_ws;
  float*  kf   = (float*)ws;                         // 8*64*1024*4 = 2 MB
  __bf16* Ktil = (__bf16*)(ws + (2u << 20));         // 1 MB
  __bf16* vT   = (__bf16*)(ws + (3u << 20));         // 1 MB
  float*  cvec = (float*)(ws + (4u << 20));          // 2 KB

  kv_kernel  <<<dim3(8, BATCH), 256, 0, stream>>>(x, Wk, bk, Wv, bv, kf, vT);
  ktil_kernel<<<dim3(8, BATCH), 256, 0, stream>>>(kf, Wq, Ktil);
  cvec_kernel<<<dim3(BATCH),     64, 0, stream>>>(kf, bq, cvec);
  fused_attn <<<dim3(SEQ / 128, BATCH), 256, 0, stream>>>(p, Ktil, vT, cvec, Wg, bg, out);
}
